// Decoder_14516989461055
// MI455X (gfx1250) — matrix operand order repaired and checked by reading, not compiled, not run on hardware
//
#include <hip/hip_runtime.h>
#include <hip/hip_bf16.h>
#include <math.h>

// ---------------------------------------------------------------------------
// Problem constants
// ---------------------------------------------------------------------------
#define B_   64
#define T_   512
#define E_   512
#define H_   256
#define P_   64
#define L_   33
#define LP_  48          // L padded to 3 WMMA N-tiles
#define KV_  768         // H + E  (input to W_lin)
#define KX_  576         // P + E  (input to W_ih)
#define FH_  1024        // 4*H
#define MB_  16          // batch rows per workgroup (= WMMA M)
#define TPB_ 256         // 8 waves

typedef __attribute__((ext_vector_type(16))) _Float16     v16h;
typedef __attribute__((ext_vector_type(8)))  float        v8f;
typedef __attribute__((ext_vector_type(8)))  unsigned int v8u;

// ---------------------------------------------------------------------------
// WMMA helpers (layouts per CDNA5 ISA §7.12.2, wave32)
// ---------------------------------------------------------------------------

// 16x16x32 f16 WMMA, f32 accumulate
__device__ __forceinline__ v8f wmma_f16(v16h a, v16h b, v8f c) {
  return __builtin_amdgcn_wmma_f32_16x16x32_f16(
      /*neg_a=*/false, a, /*neg_b=*/false, b,
      /*c_mod=*/(short)0, c, /*reuse_a=*/false, /*reuse_b=*/false);
}

// A fragment (16x32 f16, M x K): lane holds row m = lane&15.
// VGPR j holds K pair starting at: (j>=4 ? 16:0) + (lane>=16 ? 8:0) + 2*(j&3)
__device__ __forceinline__ v16h load_fragA(const _Float16* base, int stride,
                                           int kbase, int lane) {
  const _Float16* row = base + (lane & 15) * stride + kbase + ((lane & 16) ? 8 : 0);
  v8u t;
#pragma unroll
  for (int j = 0; j < 8; ++j) {
    int k0 = ((j & 4) << 2) + 2 * (j & 3);
    t[j] = *(const unsigned int*)(row + k0);     // 2 packed halves (4B aligned)
  }
  return __builtin_bit_cast(v16h, t);
}

// B fragment (32x16 f16, K x N): lane holds column n = lane&15; half s holds
// k = (s>=8 ? 16:0) + (lane>=16 ? 8:0) + (s&7), matching the A fragment.
// Weights stored transposed (k-major), so this is a strided gather.
__device__ __forceinline__ v16h load_fragB(const _Float16* WT, int nld,
                                           int kbase, int ncol0, int lane) {
  const _Float16* p = WT + (size_t)(kbase + ((lane & 16) ? 8 : 0)) * nld + ncol0 + (lane & 15);
  v16h r;
#pragma unroll
  for (int s = 0; s < 16; ++s) r[s] = p[(size_t)((s & 7) + ((s & 8) << 1)) * nld];
  return r;
}

// C/D f32 16x16: lane n = lane&15; VGPR r -> row m = r + (lane>=16 ? 8 : 0)
__device__ __forceinline__ void store_fragC(float* G, int stride, int ncol0,
                                            v8f c, int lane) {
  int n = lane & 15;
  int mb = (lane & 16) ? 8 : 0;
#pragma unroll
  for (int r = 0; r < 8; ++r) G[(mb + r) * stride + ncol0 + n] = c[r];
}

__device__ __forceinline__ float sigm(float x) { return 1.0f / (1.0f + expf(-x)); }

// ---------------------------------------------------------------------------
// Weight conversion / transposition:  f32 row-major -> f16 k-major (W^T)
//   WlinT [768][48]  (rows >= 33 zero padded)
//   WihT  [576][1024]
//   WhhT  [256][1024]
//   pos16 [32][64]    (plain)
// ---------------------------------------------------------------------------
#define NWLIN_ (KV_ * LP_)        // 36864
#define NWIH_  (KX_ * FH_)        // 589824
#define NWHH_  (H_ * FH_)         // 262144
#define NPOS_  (32 * P_)          // 2048
#define NCVT_  (NWLIN_ + NWIH_ + NWHH_ + NPOS_)

__global__ void convert_weights_kernel(const float* __restrict__ Wlin,
                                       const float* __restrict__ Wih,
                                       const float* __restrict__ Whh,
                                       const float* __restrict__ pos,
                                       _Float16* __restrict__ wlinT,
                                       _Float16* __restrict__ wihT,
                                       _Float16* __restrict__ whhT,
                                       _Float16* __restrict__ pos16) {
  for (size_t idx = blockIdx.x * blockDim.x + threadIdx.x; idx < NCVT_;
       idx += (size_t)gridDim.x * blockDim.x) {
    if (idx < NWLIN_) {
      int k = (int)(idx / LP_), n = (int)(idx % LP_);
      wlinT[idx] = (_Float16)((n < L_) ? Wlin[n * KV_ + k] : 0.0f);
    } else if (idx < NWLIN_ + NWIH_) {
      size_t j = idx - NWLIN_;
      int k = (int)(j / FH_), n = (int)(j % FH_);
      wihT[j] = (_Float16)Wih[(size_t)n * KX_ + k];
    } else if (idx < NWLIN_ + NWIH_ + NWHH_) {
      size_t j = idx - NWLIN_ - NWIH_;
      int k = (int)(j / FH_), n = (int)(j % FH_);
      whhT[j] = (_Float16)Whh[(size_t)n * H_ + k];
    } else {
      size_t j = idx - NWLIN_ - NWIH_ - NWHH_;
      pos16[j] = (_Float16)pos[j];
    }
  }
}

// ---------------------------------------------------------------------------
// Async global -> LDS copy of one encoder row tile (16 rows x 512 f32 = 32KB)
// 2048 x 16-byte chunks; each thread issues 8 async b128 transfers.
// Tracked by ASYNCcnt; retired by s_wait_asynccnt + barrier at consumption.
// ---------------------------------------------------------------------------
__device__ __forceinline__ void issue_enc_async(const float* __restrict__ enc,
                                                float* __restrict__ dstbuf,
                                                int b0, int t, int tid) {
#pragma unroll
  for (int it = 0; it < (MB_ * E_ * 4 / 16) / TPB_; ++it) {   // 8 iterations
    int ch = tid + it * TPB_;
    int b  = ch >> 7;                 // 128 chunks per 2KB row
    int e4 = (ch & 127) << 2;         // float index, multiple of 4
    const float* g = enc + ((size_t)(b0 + b) * T_ + t) * E_ + e4;
    unsigned lds = (unsigned)(uintptr_t)(dstbuf + b * E_ + e4);
    unsigned long long ga = (unsigned long long)(uintptr_t)g;
    asm volatile("global_load_async_to_lds_b128 %0, %1, off"
                 :: "v"(lds), "v"(ga) : "memory");
  }
}

__device__ __forceinline__ void wait_async_all() {
  asm volatile("s_wait_asynccnt 0x0" ::: "memory");
}

// ---------------------------------------------------------------------------
// Main sequential decoder: 1 workgroup handles 16 sequences in lockstep.
// ---------------------------------------------------------------------------
// Dynamic LDS layout:
//   sh_h     f32 [16][256]
//   sh_c     f32 [16][256]
//   sh_wsum  f32 [16][512]     running word sum (= prefix[t]-prefix[cur_ws])
//   sh_enc   f32 [2][16][512]  double-buffered async encoder tile
//   sh_G     f32 [16][1024]    LSTM gate pre-activations
//   sh_out1  f32 [16][48]      GEMM1 raw output
//   sh_invw  f32 [16]
//   sh_V     f16 [16][768]     [h | enc_t]
//   sh_X     f16 [16][576]     [pos_emb | avg]
//   8 int arrays [16]
#define SMEM_BYTES_ 243264

__global__ __launch_bounds__(TPB_, 1) void decode_kernel(
    const float* __restrict__ enc, const _Float16* __restrict__ pos16,
    const _Float16* __restrict__ WihT, const _Float16* __restrict__ WhhT,
    const _Float16* __restrict__ WlinT, const float* __restrict__ b_ih,
    const float* __restrict__ b_hh, const float* __restrict__ b_lin,
    const int* __restrict__ real_lens, float* __restrict__ out) {
  extern __shared__ char smem[];
  float*    sh_h    = (float*)smem;                  // 16*256
  float*    sh_c    = sh_h    + MB_ * H_;            // 16*256
  float*    sh_wsum = sh_c    + MB_ * H_;            // 16*512
  float*    sh_enc  = sh_wsum + MB_ * E_;            // 2*16*512
  float*    sh_G    = sh_enc  + 2 * MB_ * E_;        // 16*1024
  float*    sh_out1 = sh_G    + MB_ * FH_;           // 16*48
  float*    sh_invw = sh_out1 + MB_ * LP_;           // 16
  _Float16* sh_V    = (_Float16*)(sh_invw + MB_);    // 16*768
  _Float16* sh_X    = sh_V + MB_ * KV_;              // 16*576
  int* sh_last_sep  = (int*)(sh_X + MB_ * KX_);
  int* sh_last_pos  = sh_last_sep + MB_;
  int* sh_cur_ws    = sh_last_pos + MB_;
  int* sh_wc        = sh_cur_ws + MB_;
  int* sh_pc        = sh_wc + MB_;
  int* sh_is_sep    = sh_pc + MB_;
  int* sh_do_lstm   = sh_is_sep + MB_;
  int* sh_rl        = sh_do_lstm + MB_;

  const int tid  = threadIdx.x;
  const int wave = tid >> 5;
  const int lane = tid & 31;
  const int b0   = blockIdx.x * MB_;

  // kick off async copy of the t=0 encoder tile immediately (overlaps init
  // and the pre-loop LSTM step)
  issue_enc_async(enc, sh_enc, b0, 0, tid);

  // ---- init state -------------------------------------------------------
  for (int i = tid; i < MB_ * H_; i += TPB_) { sh_h[i] = 0.f; sh_c[i] = 0.f; }
  for (int i = tid; i < MB_ * E_; i += TPB_) sh_wsum[i] = 0.f;
  for (int i = tid; i < MB_ * KV_; i += TPB_) sh_V[i] = (_Float16)0.f;
  for (int i = tid; i < MB_ * KX_; i += TPB_) {
    int k = i % KX_;
    sh_X[i] = (k < P_) ? pos16[k] : (_Float16)0.f;   // x0 = [pos_emb[0], zeros]
  }
  if (tid < MB_) {
    sh_last_sep[tid] = 0; sh_last_pos[tid] = 0; sh_cur_ws[tid] = 0;
    sh_wc[tid] = 0; sh_pc[tid] = 0; sh_is_sep[tid] = 0;
    sh_do_lstm[tid] = 1;   // unconditional pre-loop LSTM step
    sh_rl[tid] = max(real_lens[b0 + tid], 1);
  }
  __syncthreads();

  // ---- GEMM2 lambda: G[16,1024] = X @ Wih^T + h @ Whh^T -----------------
  auto gemm2 = [&]() {
    v8f acc[8];
#pragma unroll
    for (int i = 0; i < 8; ++i) acc[i] = (v8f){0.f,0.f,0.f,0.f,0.f,0.f,0.f,0.f};
    for (int kc = 0; kc < KX_ / 32; ++kc) {          // 18 chunks over X
      v16h a = load_fragA(sh_X, KX_, kc * 32, lane);
#pragma unroll
      for (int i = 0; i < 8; ++i) {
        int nt = wave + 8 * i;
        v16h b = load_fragB(WihT, FH_, kc * 32, nt * 16, lane);
        acc[i] = wmma_f16(a, b, acc[i]);
      }
    }
    for (int kc = 0; kc < H_ / 32; ++kc) {           // 8 chunks over h (f16 in V)
      v16h a = load_fragA(sh_V, KV_, kc * 32, lane);
#pragma unroll
      for (int i = 0; i < 8; ++i) {
        int nt = wave + 8 * i;
        v16h b = load_fragB(WhhT, FH_, kc * 32, nt * 16, lane);
        acc[i] = wmma_f16(a, b, acc[i]);
      }
    }
#pragma unroll
    for (int i = 0; i < 8; ++i)
      store_fragC(sh_G, FH_, (wave + 8 * i) * 16, acc[i], lane);
  };

  // ---- pointwise LSTM cell over G ---------------------------------------
  auto lstm_pointwise = [&]() {
    for (int i = tid; i < MB_ * H_; i += TPB_) {
      int b = i >> 8, u = i & (H_ - 1);
      const float* g = sh_G + b * FH_;
      float xi = g[u]           + b_ih[u]           + b_hh[u];
      float xf = g[H_ + u]      + b_ih[H_ + u]      + b_hh[H_ + u];
      float xg = g[2 * H_ + u]  + b_ih[2 * H_ + u]  + b_hh[2 * H_ + u];
      float xo = g[3 * H_ + u]  + b_ih[3 * H_ + u]  + b_hh[3 * H_ + u];
      float c2 = sigm(xf) * sh_c[i] + sigm(xi) * tanhf(xg);
      float h2 = sigm(xo) * tanhf(c2);
      if (sh_do_lstm[b]) { sh_c[i] = c2; sh_h[i] = h2; }
    }
  };

  // ---- pre-loop LSTM step (carry0) --------------------------------------
  gemm2();
  __syncthreads();
  lstm_pointwise();
  __syncthreads();

  // ======================= sequential time loop ==========================
  for (int t = 0; t < T_; ++t) {
    const int cur = t & 1;
    float* encbuf = sh_enc + cur * (MB_ * E_);

    // retire this step's async tile; barrier makes it visible to all waves
    wait_async_all();
    __syncthreads();

    // immediately start fetching the next tile into the other buffer so the
    // HBM/L2 transfer overlaps this whole step's compute
    if (t + 1 < T_)
      issue_enc_async(enc, sh_enc + (1 - cur) * (MB_ * E_), b0, t + 1, tid);

    // phase F: rebuild V = [h | enc_t] in f16 (pure LDS traffic now)
    for (int i = tid; i < MB_ * E_; i += TPB_) {
      int b = i >> 9, e = i & (E_ - 1);
      sh_V[b * KV_ + H_ + e] = (_Float16)encbuf[i];
    }
    for (int i = tid; i < MB_ * H_; i += TPB_) {
      int b = i >> 8, k = i & (H_ - 1);
      sh_V[b * KV_ + k] = (_Float16)sh_h[i];
    }
    __syncthreads();

    // phase A: GEMM1  out1[16,48] = V[16,768] @ Wlin^T   (waves 0..2)
    if (wave < 3) {
      v8f acc = (v8f){0.f,0.f,0.f,0.f,0.f,0.f,0.f,0.f};
      for (int kc = 0; kc < KV_ / 32; ++kc) {
        v16h a = load_fragA(sh_V, KV_, kc * 32, lane);
        v16h b = load_fragB(WlinT, LP_, kc * 32, wave * 16, lane);
        acc = wmma_f16(a, b, acc);
      }
      store_fragC(sh_out1, LP_, wave * 16, acc, lane);
    }
    __syncthreads();

    // phase B: masked tanh outputs + per-sequence argmax/state machine
    for (int i = tid; i < MB_ * L_; i += TPB_) {
      int b = i / L_, l = i - b * L_;
      float val = (t < sh_rl[b]) ? tanhf(sh_out1[b * LP_ + l] + b_lin[l]) : 0.f;
      out[((size_t)(b0 + b) * T_ + t) * L_ + l] = val;
    }
    if (tid < MB_) {
      int b = tid;
      int valid = (t < sh_rl[b]);
      float best = -INFINITY; int a = 0;
      for (int l = 0; l < L_; ++l) {                 // tanh monotone: raw argmax
        float v = sh_out1[b * LP_ + l] + b_lin[l];
        if (v > best) { best = v; a = l; }
      }
      int is_sep = (a > 0) && valid;
      int last_sep = sh_last_sep[b], last_pos = sh_last_pos[b];
      int cur_ws = sh_cur_ws[b], wc = sh_wc[b], pc = sh_pc[b];
      int last_pos_new = (is_sep && pc >= 1) ? last_sep : last_pos;
      sh_last_sep[b] = is_sep ? (a - 1) : last_sep;
      sh_last_pos[b] = last_pos_new;
      sh_pc[b]       = pc + is_sep;
      sh_wc[b]       = valid ? (is_sep ? wc + 1 : max(wc, 1)) : wc;
      sh_do_lstm[b]  = is_sep && (wc >= 1);
      sh_is_sep[b]   = is_sep;
      sh_invw[b]     = 1.0f / (float)max(t - cur_ws, 1);
      sh_cur_ws[b]   = is_sep ? t : cur_ws;
    }
    __syncthreads();

    // phase C: X = [pos_emb[last_pos_new] | wsum/wlen] in f16
    for (int i = tid; i < MB_ * KX_; i += TPB_) {
      int b = i / KX_, k = i - b * KX_;
      _Float16 v;
      if (k < P_) v = pos16[sh_last_pos[b] * P_ + k];
      else        v = (_Float16)(sh_wsum[b * E_ + (k - P_)] * sh_invw[b]);
      sh_X[i] = v;
    }
    __syncthreads();

    // phase D: GEMM2 (all 8 waves, 208 WMMA each)
    gemm2();
    __syncthreads();

    // phase E: conditional LSTM state update + running word-sum update
    lstm_pointwise();
    for (int i = tid; i < MB_ * E_; i += TPB_) {
      int b = i >> 9;
      float w = sh_is_sep[b] ? 0.f : sh_wsum[i];
      sh_wsum[i] = w + encbuf[i];
    }
    __syncthreads();
  }
}

// ---------------------------------------------------------------------------
// log_softmax over last dim (L=33), one wave32 per row, in-place on d_out
// ---------------------------------------------------------------------------
__global__ __launch_bounds__(TPB_) void logsoftmax_kernel(float* __restrict__ out) {
  int row  = blockIdx.x * (TPB_ / 32) + (threadIdx.x >> 5);
  int lane = threadIdx.x & 31;
  if (row >= B_ * T_) return;
  float* p = out + (size_t)row * L_;
  float x0 = p[lane];
  float x1 = (lane == 0) ? p[32] : -INFINITY;
  float m = fmaxf(x0, x1);
  for (int o = 16; o > 0; o >>= 1) m = fmaxf(m, __shfl_xor(m, o, 32));
  float s = expf(x0 - m) + expf(x1 - m);
  for (int o = 16; o > 0; o >>= 1) s += __shfl_xor(s, o, 32);
  float lse = m + logf(s);
  p[lane] = x0 - lse;
  if (lane == 0) p[32] = x1 - lse;
}

// ---------------------------------------------------------------------------
// Host launcher
// ---------------------------------------------------------------------------
extern "C" void kernel_launch(void* const* d_in, const int* in_sizes, int n_in,
                              void* d_out, int out_size, void* d_ws,
                              size_t ws_size, hipStream_t stream) {
  (void)in_sizes; (void)n_in; (void)out_size; (void)ws_size;
  const float* enc  = (const float*)d_in[0];
  const float* pos  = (const float*)d_in[1];
  const float* Wih  = (const float*)d_in[2];
  const float* Whh  = (const float*)d_in[3];
  const float* bih  = (const float*)d_in[4];
  const float* bhh  = (const float*)d_in[5];
  const float* Wlin = (const float*)d_in[6];
  const float* blin = (const float*)d_in[7];
  const int*   rlen = (const int*)d_in[8];
  float*       outp = (float*)d_out;

  // workspace layout (256B-aligned regions), ~1.70 MB total
  char* ws = (char*)d_ws;
  _Float16* wlinT = (_Float16*)(ws + 0);                         // 48*768*2   = 73728
  _Float16* wihT  = (_Float16*)(ws + 73728);                     // 576*1024*2 = 1179648
  _Float16* whhT  = (_Float16*)(ws + 73728 + 1179648);           // 256*1024*2 = 524288
  _Float16* pos16 = (_Float16*)(ws + 73728 + 1179648 + 524288);  // 32*64*2    = 4096

  convert_weights_kernel<<<1024, TPB_, 0, stream>>>(Wlin, Wih, Whh, pos,
                                                    wlinT, wihT, whhT, pos16);

  hipFuncSetAttribute(reinterpret_cast<const void*>(decode_kernel),
                      hipFuncAttributeMaxDynamicSharedMemorySize, SMEM_BYTES_);
  decode_kernel<<<B_ / MB_, TPB_, SMEM_BYTES_, stream>>>(
      enc, pos16, wihT, whhT, wlinT, bih, bhh, blin, rlen, outp);

  logsoftmax_kernel<<<(B_ * T_) / (TPB_ / 32), TPB_, 0, stream>>>(outp);
}
